// SelfAttn_25744033972336
// MI455X (gfx1250) — matrix operand order repaired and checked by reading, not compiled, not run on hardware
//
#include <hip/hip_runtime.h>
#include <hip/hip_bf16.h>

// ---------------------------------------------------------------------------
// Self-attention (SAGAN-style) for MI455X / gfx1250.
// - All GEMMs via V_WMMA_F32_16X16X32_BF16 (wave32).
// - Flash-style softmax: never materializes the 4096x2048 score matrix.
// - Attention streams phi/g m-chunks into LDS via the Tensor Data Mover
//   (tensor_load_to_lds + s_wait_tensorcnt), double-buffered so DMA overlaps
//   WMMA compute. phi/g per batch (1.25MB) stay L2-resident; TDM pulls L2->LDS.
// ---------------------------------------------------------------------------

typedef __attribute__((ext_vector_type(16))) __bf16 v16bf;
typedef __attribute__((ext_vector_type(8)))  float  v8f;
typedef __attribute__((ext_vector_type(4)))  unsigned int v4u;
typedef __attribute__((ext_vector_type(8)))  int v8i;
typedef __attribute__((ext_vector_type(4)))  int v4i;

#define DEVINL __device__ __forceinline__

constexpr int BB    = 8;
constexpr int CH    = 512;
constexpr int LL    = 4096;
constexpr int CTH   = 64;    // ch/8
constexpr int CG    = 256;   // ch/2
constexpr int MM    = 2048;  // L/2 (after maxpool2)
constexpr int MROWS = 384;   // 64 + 64 + 256 concatenated projection rows

// ---- workspace layout (bytes) ----
constexpr size_t OFF_WCAT = 0;
constexpr size_t SZ_WCAT  = (size_t)MROWS * CH * 2;
constexpr size_t OFF_WO   = OFF_WCAT + SZ_WCAT;
constexpr size_t SZ_WO    = (size_t)CH * CG * 2;
constexpr size_t OFF_THT  = OFF_WO + SZ_WO;                 // thetaT: (B, L, 64) bf16
constexpr size_t SZ_THT   = (size_t)BB * LL * CTH * 2;
constexpr size_t OFF_PHI  = OFF_THT + SZ_THT;               // phi:    (B, 64, 2048) bf16
constexpr size_t SZ_PHI   = (size_t)BB * CTH * MM * 2;
constexpr size_t OFF_G    = OFF_PHI + SZ_PHI;               // g:      (B, 256, 2048) bf16
constexpr size_t SZ_G     = (size_t)BB * CG * MM * 2;
constexpr size_t OFF_OPT  = OFF_G + SZ_G;                   // o_preT: (B, L, 256) bf16

DEVINL v8f vzero() {
  v8f v;
#pragma unroll
  for (int i = 0; i < 8; ++i) v[i] = 0.f;
  return v;
}

DEVINL v8f wmma_bf16(v16bf a, v16bf b, v8f c) {
  return __builtin_amdgcn_wmma_f32_16x16x32_bf16(
      /*neg_a=*/false, a, /*neg_b=*/false, b,
      /*c_mod=*/(short)0, c, /*reuse_a=*/false, /*reuse_b=*/false);
}

// A fragment: row-major bf16, lane holds row m = m_base + (lane&15),
// element e -> K = k_base + half*16 + e
DEVINL v16bf load_frag_A(const __bf16* A, int m_base, int k_base, int lda, int lane) {
  const int m  = m_base + (lane & 15);
  const int hf = lane >> 4;
  const __bf16* p = A + (size_t)m * lda + k_base + hf * 16;
  v16bf v;
#pragma unroll
  for (int e = 0; e < 8; ++e) v[e] = p[e];
#pragma unroll
  for (int e = 0; e < 8; ++e) v[8 + e] = p[8 + e];
  return v;
}

// B fragment: element (k,n) at B[k*ldb + n]; lane holds column n,
// element e -> K = k_base + half*16 + e   (contiguous k order, same as load_frag_A)
DEVINL v16bf load_frag_Brm(const __bf16* B, int k_base, int n_base, int ldb, int lane) {
  const int n  = n_base + (lane & 15);
  const int hf = lane >> 4;
  const __bf16* p = B + (size_t)(k_base + hf * 16) * ldb + n;
  v16bf v;
#pragma unroll
  for (int e = 0; e < 16; ++e) v[e] = p[(size_t)e * ldb];
  return v;
}

// B fragment from fp32 row-major source with on-the-fly bf16 conversion
DEVINL v16bf load_frag_B_f32(const float* B, int k_base, int n_base, int ldb, int lane) {
  const int n  = n_base + (lane & 15);
  const int hf = lane >> 4;
  const float* p = B + (size_t)(k_base + hf * 16) * ldb + n;
  v16bf v;
#pragma unroll
  for (int e = 0; e < 16; ++e) v[e] = (__bf16)p[(size_t)e * ldb];
  return v;
}

// B fragment, column-major storage: element (k,n) at B[n*ldc + k]
DEVINL v16bf load_frag_Bcm(const __bf16* B, int k_base, int n_base, int ldc, int lane) {
  const int n  = n_base + (lane & 15);
  const int hf = lane >> 4;
  const __bf16* p = B + (size_t)n * ldc + k_base + hf * 16;
  v16bf v;
#pragma unroll
  for (int e = 0; e < 16; ++e) v[e] = p[e];
  return v;
}

// ---------------------------------------------------------------------------
// Tensor Data Mover: 2D tile (bf16) global -> LDS.
// D# packing per CDNA5 ISA:
//   g0: [1:0]=count=1, [63:32]=lds_addr, [120:64]=global_addr, [127:126]=2
//   g1: [17:16]=data_size(1=2B), [79:48]=tensor_dim0, [111:80]=tensor_dim1,
//       [127:112]=tile_dim0, [143:128]=tile_dim1, [207:160]=dim0_stride
// ---------------------------------------------------------------------------
DEVINL void tdm_load_2d_bf16(const void* gsrc, unsigned lds_off,
                             unsigned tile_d0, unsigned tile_d1,
                             unsigned tens_d0, unsigned tens_d1,
                             unsigned stride0_elems) {
  const unsigned long long ga = (unsigned long long)(uintptr_t)gsrc;
  v4u g0;
  g0[0] = 1u;                                            // count=1 (valid)
  g0[1] = lds_off;                                       // lds_addr (bytes)
  g0[2] = (unsigned)(ga & 0xFFFFFFFFu);                  // global_addr lo
  g0[3] = (unsigned)((ga >> 32) & 0x01FFFFFFu) | (2u << 30);  // hi + type=2
  v8i g1;
  g1[0] = (int)(1u << 16);                               // data_size = 2 bytes
  g1[1] = (int)((tens_d0 & 0xFFFFu) << 16);              // tensor_dim0 lo16
  g1[2] = (int)((tens_d0 >> 16) | ((tens_d1 & 0xFFFFu) << 16));
  g1[3] = (int)((tens_d1 >> 16) | ((tile_d0 & 0xFFFFu) << 16));
  g1[4] = (int)(tile_d1 & 0xFFFFu);                      // tile_dim1; tile_dim2=0
  g1[5] = (int)stride0_elems;                            // dim0 stride lo32
  g1[6] = 0;
  g1[7] = 0;
  v4i gz = {0, 0, 0, 0};
#if defined(__clang_major__) && (__clang_major__ >= 23)
  v8i gz8 = {0, 0, 0, 0, 0, 0, 0, 0};
  __builtin_amdgcn_tensor_load_to_lds(g0, g1, gz, gz, gz8, 0);
#else
  __builtin_amdgcn_tensor_load_to_lds(g0, g1, gz, gz, 0);
#endif
}

// ---------------------------------------------------------------------------
// Kernel 1: weight fp32 -> bf16 (Wcat = [W_theta; W_phi; W_g], Wo)
// ---------------------------------------------------------------------------
__global__ void k_convert(const float* Wth, const float* Wph, const float* Wg,
                          const float* Wof, __bf16* Wcat, __bf16* Wo) {
  const int i = blockIdx.x * 256 + threadIdx.x;
  constexpr int S1 = 64 * 512, S2 = 128 * 512, S3 = 384 * 512;
  if (i < S1)                Wcat[i] = (__bf16)Wth[i];
  else if (i < S2)           Wcat[i] = (__bf16)Wph[i - S1];
  else if (i < S3)           Wcat[i] = (__bf16)Wg[i - S2];
  else if (i < S3 + CH * CG) Wo[i - S3] = (__bf16)Wof[i - S3];
}

// ---------------------------------------------------------------------------
// Kernel 2: projections P = Wcat(384x512) @ x_b(512x4096); epilogue writes
// thetaT (transposed) and maxpool2'd phi / g via lane-pair shfl max.
// grid = B * 24 mtiles * 16 nstrips(256 l); block = 128 (4 waves x 64 l)
// ---------------------------------------------------------------------------
__global__ void k_proj(const __bf16* __restrict__ Wcat, const float* __restrict__ x,
                       __bf16* __restrict__ thetaT, __bf16* __restrict__ phi,
                       __bf16* __restrict__ g) {
  const int tid  = threadIdx.x;
  const int w    = tid >> 5;
  const int lane = tid & 31;
  const int n15  = lane & 15;
  const int hf   = lane >> 4;

  int bs = blockIdx.x;
  const int ns = bs & 15;
  bs >>= 4;
  const int mt = bs % 24;
  const int b  = bs / 24;

  const float* xb = x + (size_t)b * CH * LL;
  __bf16* thT = thetaT + (size_t)b * LL * CTH;
  __bf16* ph  = phi    + (size_t)b * CTH * MM;
  __bf16* gg  = g      + (size_t)b * CG * MM;

  const int l_base = ns * 256 + w * 64;

  v8f acc[4];
#pragma unroll
  for (int t = 0; t < 4; ++t) acc[t] = vzero();

  for (int ks = 0; ks < CH; ks += 32) {
    if (ks + 32 < CH)
      __builtin_prefetch(xb + (size_t)(ks + 32 + hf * 16) * LL + l_base + n15 * 4, 0, 0);
    v16bf a = load_frag_A(Wcat, mt * 16, ks, CH, lane);
#pragma unroll
    for (int t = 0; t < 4; ++t) {
      v16bf bf = load_frag_B_f32(xb, ks, l_base + t * 16, LL, lane);
      acc[t] = wmma_bf16(a, bf, acc[t]);
    }
  }

#pragma unroll
  for (int t = 0; t < 4; ++t) {
    const int n = l_base + t * 16 + n15;
#pragma unroll
    for (int r = 0; r < 8; ++r) {
      const int m = mt * 16 + r + hf * 8;
      const float v = acc[t][r];
      if (mt < 4) {
        thT[(size_t)n * CTH + m] = (__bf16)v;
      } else {
        const float other = __shfl_xor(v, 1, 32);
        const float mx = fmaxf(v, other);
        if ((n & 1) == 0) {
          if (mt < 8) ph[(size_t)(m - 64) * MM + (n >> 1)]  = (__bf16)mx;
          else        gg[(size_t)(m - 128) * MM + (n >> 1)] = (__bf16)mx;
        }
      }
    }
  }
}

// ---------------------------------------------------------------------------
// Kernel 3: flash attention with TDM-staged phi/g tiles.
// Per block: one 16-row l-tile; 4 waves cover c=0..255 (64 each).
// m streamed in chunks of 32; wave 0 drives double-buffered TDM loads:
//   sphi tile: 64 x 32 bf16 (4KB)   [c*32 + mlocal]
//   sg   tile: 256 x 32 bf16 (16KB) [c*32 + mlocal]
// grid = B * 256 l-tiles; block = 128
// ---------------------------------------------------------------------------
__global__ void k_attn(const __bf16* __restrict__ thetaT, const __bf16* __restrict__ phi,
                       const __bf16* __restrict__ g, __bf16* __restrict__ o_preT) {
  __shared__ __align__(128) __bf16 sphi[2][CTH * 32];
  __shared__ __align__(128) __bf16 sg[2][CG * 32];
  __shared__ __bf16 Plds[4][16 * 34];

  const int tid  = threadIdx.x;
  const int w    = tid >> 5;
  const int lane = tid & 31;
  const int n15  = lane & 15;
  const int hf   = lane >> 4;

  const int bs = blockIdx.x;
  const int lt = bs & 255;
  const int b  = bs >> 8;

  const __bf16* thT = thetaT + (size_t)b * LL * CTH;
  const __bf16* ph  = phi    + (size_t)b * CTH * MM;
  const __bf16* gg  = g      + (size_t)b * CG * MM;
  __bf16* op = o_preT + (size_t)b * LL * CG;

  const int c_base = w * 64;
  constexpr int NCHUNK = MM / 32;

  const v16bf a0 = load_frag_A(thT, lt * 16, 0, CTH, lane);
  const v16bf a1 = load_frag_A(thT, lt * 16, 32, CTH, lane);

  v8f acc[4];
#pragma unroll
  for (int cc = 0; cc < 4; ++cc) acc[cc] = vzero();
  float rmax[8], rsum[8];
#pragma unroll
  for (int r = 0; r < 8; ++r) { rmax[r] = -3.0e38f; rsum[r] = 0.f; }

  // prologue: TDM-load chunk 0 into buffer 0 (wave 0 only; TENSORcnt is per-wave)
  if (w == 0) {
    tdm_load_2d_bf16(ph, (unsigned)(uintptr_t)&sphi[0][0], 32, CTH, MM, CTH, MM);
    tdm_load_2d_bf16(gg, (unsigned)(uintptr_t)&sg[0][0],   32, CG,  MM, CG,  MM);
  }

  for (int it = 0; it < NCHUNK; ++it) {
    const int cur = it & 1;
    if (w == 0) {
      if (it + 1 < NCHUNK) {
        const int m1 = (it + 1) * 32;
        tdm_load_2d_bf16(ph + m1, (unsigned)(uintptr_t)&sphi[cur ^ 1][0], 32, CTH, MM, CTH, MM);
        tdm_load_2d_bf16(gg + m1, (unsigned)(uintptr_t)&sg[cur ^ 1][0],   32, CG,  MM, CG,  MM);
        __builtin_amdgcn_s_wait_tensorcnt(2);  // chunk `it` done; next chunk in flight
      } else {
        __builtin_amdgcn_s_wait_tensorcnt(0);
      }
    }
    __syncthreads();  // chunk `it` visible to all waves

    const __bf16* sphb = &sphi[cur][0];
    const __bf16* sgb  = &sg[cur][0];

    // S = theta^T @ phi : two 16x16 tiles (local columns 0..15, 16..31)
    v8f s0 = vzero(), s1 = vzero();
    s0 = wmma_bf16(a0, load_frag_Brm(sphb, 0,  0,  32, lane), s0);
    s0 = wmma_bf16(a1, load_frag_Brm(sphb, 32, 0,  32, lane), s0);
    s1 = wmma_bf16(a0, load_frag_Brm(sphb, 0,  16, 32, lane), s1);
    s1 = wmma_bf16(a1, load_frag_Brm(sphb, 32, 16, 32, lane), s1);

    // online softmax: row r+hf*8 lives within one 16-lane half
    float sc[8];
#pragma unroll
    for (int r = 0; r < 8; ++r) {
      float v = fmaxf(s0[r], s1[r]);
      v = fmaxf(v, __shfl_xor(v, 1, 16));
      v = fmaxf(v, __shfl_xor(v, 2, 16));
      v = fmaxf(v, __shfl_xor(v, 4, 16));
      v = fmaxf(v, __shfl_xor(v, 8, 16));
      const float nm = fmaxf(rmax[r], v);
      sc[r] = __expf(rmax[r] - nm);
      rmax[r] = nm;
      const float p0 = __expf(s0[r] - nm);
      const float p1 = __expf(s1[r] - nm);
      float ps = p0 + p1;
      ps += __shfl_xor(ps, 1, 16);
      ps += __shfl_xor(ps, 2, 16);
      ps += __shfl_xor(ps, 4, 16);
      ps += __shfl_xor(ps, 8, 16);
      rsum[r] = rsum[r] * sc[r] + ps;
      const int row = r + hf * 8;
      Plds[w][row * 34 + n15]      = (__bf16)p0;
      Plds[w][row * 34 + 16 + n15] = (__bf16)p1;
    }
#pragma unroll
    for (int cc = 0; cc < 4; ++cc)
#pragma unroll
      for (int r = 0; r < 8; ++r) acc[cc][r] *= sc[r];

    __syncthreads();
    // P tile: D-layout -> A-layout via LDS
    v16bf a2;
    {
      const __bf16* p = &Plds[w][n15 * 34];
#pragma unroll
      for (int e = 0; e < 16; ++e) {
        const int k = hf * 16 + e;
        a2[e] = p[k];
      }
    }

    // acc += P(16x32) @ g^T ; sg stores (c, mlocal) -> column-major B, contiguous
#pragma unroll
    for (int cc = 0; cc < 4; ++cc) {
      v16bf bg = load_frag_Bcm(sgb, 0, c_base + cc * 16, 32, lane);
      acc[cc] = wmma_bf16(a2, bg, acc[cc]);
    }
    __syncthreads();  // release buffer `cur` for chunk it+2; also covers Plds reuse
  }

#pragma unroll
  for (int cc = 0; cc < 4; ++cc) {
    const int c = c_base + cc * 16 + n15;
#pragma unroll
    for (int r = 0; r < 8; ++r) {
      const int l = lt * 16 + r + hf * 8;
      op[(size_t)l * CG + c] = (__bf16)(acc[cc][r] / rsum[r]);
    }
  }
}

// ---------------------------------------------------------------------------
// Kernel 4: out = gamma * (Wo(512x256) @ o_pre) + x
// grid = B * 32 mtiles * 16 nstrips; block = 128 (4 waves x 64 l)
// ---------------------------------------------------------------------------
__global__ void k_out(const __bf16* __restrict__ Wo, const __bf16* __restrict__ o_preT,
                      const float* __restrict__ x, const float* __restrict__ gamma,
                      float* __restrict__ out) {
  const int tid  = threadIdx.x;
  const int w    = tid >> 5;
  const int lane = tid & 31;
  const int n15  = lane & 15;
  const int hf   = lane >> 4;

  int bs = blockIdx.x;
  const int ns = bs & 15;
  bs >>= 4;
  const int mt = bs & 31;
  const int b  = bs >> 5;

  const __bf16* op = o_preT + (size_t)b * LL * CG;
  const float*  xb = x   + (size_t)b * CH * LL;
  float*        ob = out + (size_t)b * CH * LL;

  const int l_base = ns * 256 + w * 64;
  const float gm = gamma[0];

  v8f acc[4];
#pragma unroll
  for (int t = 0; t < 4; ++t) acc[t] = vzero();

  for (int ks = 0; ks < CG; ks += 32) {
    v16bf a = load_frag_A(Wo, mt * 16, ks, CG, lane);
#pragma unroll
    for (int t = 0; t < 4; ++t) {
      v16bf bf = load_frag_Bcm(op, ks, l_base + t * 16, CG, lane);
      acc[t] = wmma_bf16(a, bf, acc[t]);
    }
  }

#pragma unroll
  for (int t = 0; t < 4; ++t) {
    const int l = l_base + t * 16 + n15;
#pragma unroll
    for (int r = 0; r < 8; ++r) {
      const int m = mt * 16 + r + hf * 8;
      const size_t idx = (size_t)m * LL + l;
      ob[idx] = gm * acc[t][r] + xb[idx];
    }
  }
}

// ---------------------------------------------------------------------------
extern "C" void kernel_launch(void* const* d_in, const int* in_sizes, int n_in,
                              void* d_out, int out_size, void* d_ws, size_t ws_size,
                              hipStream_t stream) {
  const float* x     = (const float*)d_in[0];
  const float* Wth   = (const float*)d_in[1];
  const float* Wph   = (const float*)d_in[2];
  const float* Wg    = (const float*)d_in[3];
  const float* Wof   = (const float*)d_in[4];
  const float* gamma = (const float*)d_in[5];
  float* out = (float*)d_out;

  char* ws = (char*)d_ws;
  __bf16* Wcat   = (__bf16*)(ws + OFF_WCAT);
  __bf16* Wo     = (__bf16*)(ws + OFF_WO);
  __bf16* thetaT = (__bf16*)(ws + OFF_THT);
  __bf16* phi    = (__bf16*)(ws + OFF_PHI);
  __bf16* g      = (__bf16*)(ws + OFF_G);
  __bf16* o_preT = (__bf16*)(ws + OFF_OPT);

  k_convert<<<1280, 256, 0, stream>>>(Wth, Wph, Wg, Wof, Wcat, Wo);
  k_proj<<<BB * 24 * 16, 128, 0, stream>>>(Wcat, x, thetaT, phi, g);
  k_attn<<<BB * 256, 128, 0, stream>>>(thetaT, phi, g, o_preT);
  k_out<<<BB * 32 * 16, 128, 0, stream>>>(Wo, o_preT, x, gamma, out);
}
